// MSARowAttentionWithPairBias_50903952392539
// MI455X (gfx1250) — matrix operand order repaired and checked by reading, not compiled, not run on hardware
//
#include <hip/hip_runtime.h>
#include <hip/hip_bf16.h>
#include <math.h>

typedef __attribute__((ext_vector_type(16))) _Float16 v16h;
typedef __attribute__((ext_vector_type(8)))  _Float16 v8h;
typedef __attribute__((ext_vector_type(8)))  float    v8f;
typedef __attribute__((ext_vector_type(4)))  float    v4f;

static __device__ inline float wred_sum(float v) {
#pragma unroll
  for (int o = 16; o > 0; o >>= 1) v += __shfl_xor(v, o, 32);
  return v;
}
static __device__ inline float wred_max(float v) {
#pragma unroll
  for (int o = 16; o > 0; o >>= 1) v = fmaxf(v, __shfl_xor(v, o, 32));
  return v;
}

// 16-bit A-matrix 16x32 fragment (CDNA5 ISA): lanes 0-15 kb=0, lanes 16-31 kb=8;
// per lane halves {kb..kb+7, kb+16..kb+23} of row m=lane&15.
static __device__ inline v16h load_frag_a(const _Float16* __restrict__ base, int ld, int lane) {
  int m  = lane & 15;
  int kb = (lane >> 4) << 3;
  const _Float16* p = base + m * ld + kb;
  v8h lo = *(const v8h*)(p);
  v8h hi = *(const v8h*)(p + 16);
  return __builtin_shufflevector(lo, hi, 0,1,2,3,4,5,6,7,8,9,10,11,12,13,14,15);
}
// 16-bit B-matrix 32x16 fragment, operand stored n-major ([n][k]): lane n=lane&15,
// lanes 0-15 kb=0, lanes 16-31 kb=8; per lane halves {kb..kb+7, kb+16..kb+23}.
static __device__ inline v16h load_frag_b(const _Float16* __restrict__ base, int ld, int lane) {
  int n  = lane & 15;
  int kb = (lane >> 4) << 3;
  const _Float16* p = base + n * ld + kb;
  v8h lo = *(const v8h*)(p);
  v8h hi = *(const v8h*)(p + 16);
  return __builtin_shufflevector(lo, hi, 0,1,2,3,4,5,6,7,8,9,10,11,12,13,14,15);
}
static __device__ inline v8f wmma32(v16h a, v16h b, v8f c) {
  return __builtin_amdgcn_wmma_f32_16x16x32_f16(false, a, false, b, (short)0, c, false, false);
}

// CDNA5 async global->LDS copy (ASYNCcnt-tracked), 16B per lane.
// lds_off: byte offset relative to LDS base (low 32 bits of a generic LDS pointer).
static __device__ inline void async_copy_b128(unsigned lds_off, const _Float16* g) {
  asm volatile("global_load_async_to_lds_b128 %0, %1, off"
               :: "v"(lds_off), "v"(g) : "memory");
}
#define WAIT_ASYNC(imm) asm volatile("s_wait_asynccnt " #imm ::: "memory")

// ---------------------------------------------------------------------------
// Kernel 1: weights -> f16, transposed to NxK; 1/sqrt(32) folded into Wq.
// ---------------------------------------------------------------------------
__global__ void prep_weights_kernel(const float* __restrict__ Wq, const float* __restrict__ Wk,
                                    const float* __restrict__ Wv, const float* __restrict__ Wg,
                                    const float* __restrict__ Wo,
                                    _Float16* __restrict__ WT, _Float16* __restrict__ WoT) {
  int idx = blockIdx.x * 256 + threadIdx.x;
  if (idx < 1024 * 256) {
    int j = idx >> 8, kk = idx & 255;
    int proj = j >> 8, jc = j & 255;
    const float* W = (proj == 0) ? Wq : (proj == 1) ? Wk : (proj == 2) ? Wv : Wg;
    float val = W[kk * 256 + jc];
    if (proj == 0) val *= 0.1767766952966369f;
    WT[(size_t)j * 256 + kk] = (_Float16)val;
  } else {
    int t = idx - 1024 * 256;
    if (t < 256 * 256) {
      int j = t >> 8, kk = t & 255;
      WoT[(size_t)j * 256 + kk] = (_Float16)Wo[kk * 256 + j];
    }
  }
}

// ---------------------------------------------------------------------------
// Kernel 2: LayerNorm(m) -> f16, one wave per row of 256
// ---------------------------------------------------------------------------
__global__ __launch_bounds__(256) void ln_m_kernel(const float* __restrict__ m,
                                                   const float* __restrict__ w,
                                                   const float* __restrict__ b,
                                                   _Float16* __restrict__ out) {
  int wave = threadIdx.x >> 5, lane = threadIdx.x & 31;
  int row = blockIdx.x * 8 + wave;
  const float* src = m + (size_t)row * 256 + lane * 8;
  v4f x0 = *(const v4f*)src;
  v4f x1 = *(const v4f*)(src + 4);
  float s = 0.f, s2 = 0.f;
#pragma unroll
  for (int i = 0; i < 4; ++i) { s += x0[i] + x1[i]; s2 += x0[i]*x0[i] + x1[i]*x1[i]; }
  s = wred_sum(s); s2 = wred_sum(s2);
  float mu = s * (1.f / 256.f);
  float rs = rsqrtf(s2 * (1.f / 256.f) - mu * mu + 1e-5f);
  union { v8h v; _Float16 e[8]; } o;
#pragma unroll
  for (int i = 0; i < 4; ++i) {
    o.e[i]     = (_Float16)((x0[i] - mu) * rs * w[lane*8 + i]     + b[lane*8 + i]);
    o.e[i + 4] = (_Float16)((x1[i] - mu) * rs * w[lane*8 + 4 + i] + b[lane*8 + 4 + i]);
  }
  *(v8h*)(out + (size_t)row * 256 + lane * 8) = o.v;
}

// ---------------------------------------------------------------------------
// Kernel 3: LayerNorm(z) fused with pair-bias einsum -> bias[h][q][k] (f32)
// ---------------------------------------------------------------------------
__global__ __launch_bounds__(256) void ln_z_bias_kernel(const float* __restrict__ z,
                                                        const float* __restrict__ w,
                                                        const float* __restrict__ b,
                                                        const float* __restrict__ Wz,
                                                        float* __restrict__ bias) {
  int wave = threadIdx.x >> 5, lane = threadIdx.x & 31;
  int row = blockIdx.x * 8 + wave;
  v4f x = *(const v4f*)(z + (size_t)row * 128 + lane * 4);
  float s = 0.f, s2 = 0.f;
#pragma unroll
  for (int i = 0; i < 4; ++i) { s += x[i]; s2 += x[i] * x[i]; }
  s = wred_sum(s); s2 = wred_sum(s2);
  float mu = s * (1.f / 128.f);
  float rs = rsqrtf(s2 * (1.f / 128.f) - mu * mu + 1e-5f);
  float zn[4];
#pragma unroll
  for (int i = 0; i < 4; ++i) zn[i] = (x[i] - mu) * rs * w[lane*4 + i] + b[lane*4 + i];
  float outv = 0.f;
#pragma unroll
  for (int h = 0; h < 8; ++h) {
    float a = 0.f;
#pragma unroll
    for (int i = 0; i < 4; ++i) a += zn[i] * Wz[(lane*4 + i) * 8 + h];
    a = wred_sum(a);
    if (lane == h) outv = a;
  }
  if (lane < 8) bias[(size_t)lane * 65536 + row] = outv;
}

// ---------------------------------------------------------------------------
// Kernel 4: q/k/v/g projection GEMM (65536x256)x(256x1024).
// BM=128, BN=64, BK=32; 8 waves in 4x2 grid, each owns a 2x2 subtile block
// (4 WMMA per k-step from 2 A-frags x 2 B-frags). Double-buffered LDS fed by
// global_load_async_to_lds_b128 (ASYNCcnt), overlapping DMA with WMMA.
// ---------------------------------------------------------------------------
__global__ __launch_bounds__(256) void gemm_qkvg_kernel(const _Float16* __restrict__ A,
                                                        const _Float16* __restrict__ BT,
                                                        const float* __restrict__ bg,
                                                        _Float16* __restrict__ qbuf,
                                                        _Float16* __restrict__ kbuf,
                                                        _Float16* __restrict__ vbuf,
                                                        _Float16* __restrict__ gbuf) {
  __shared__ __align__(32) _Float16 sA[2][128 * 32];
  __shared__ __align__(32) _Float16 sB[2][64 * 32];
  int tid = threadIdx.x, lane = tid & 31, wv = tid >> 5;
  int ibase = blockIdx.y * 128, jbase = blockIdx.x * 64;
  unsigned aBase = (unsigned)(size_t)(&sA[0][0]);
  unsigned bBase = (unsigned)(size_t)(&sB[0][0]);

  auto issue_tile = [&](int kt, int buf) {
    int k0 = kt * 32;
#pragma unroll
    for (int j = 0; j < 2; ++j) {                     // A: 128x32 = 512 chunks, 2/thread
      int c = tid * 2 + j, row = c >> 2, off = (c & 3) * 8;
      async_copy_b128(aBase + (unsigned)(buf * 128 * 32 + row * 32 + off) * 2,
                      A + (size_t)(ibase + row) * 256 + k0 + off);
    }
    int n = tid >> 2, off = (tid & 3) * 8;            // B: 64x32 = 256 chunks, 1/thread
    async_copy_b128(bBase + (unsigned)(buf * 64 * 32 + n * 32 + off) * 2,
                    BT + (size_t)(jbase + n) * 256 + k0 + off);
  };

  int mtb = (wv & 3) * 2, ntb = (wv >> 2) * 2;
  v8f acc00 = {}, acc01 = {}, acc10 = {}, acc11 = {};
  issue_tile(0, 0);
  for (int kt = 0; kt < 8; ++kt) {
    int buf = kt & 1;
    if (kt < 7) { issue_tile(kt + 1, buf ^ 1); WAIT_ASYNC(0x3); }
    else        { WAIT_ASYNC(0x0); }
    __syncthreads();
    v16h a0 = load_frag_a(&sA[buf][(mtb    ) * 16 * 32], 32, lane);
    v16h a1 = load_frag_a(&sA[buf][(mtb + 1) * 16 * 32], 32, lane);
    v16h b0 = load_frag_b(&sB[buf][(ntb    ) * 16 * 32], 32, lane);
    v16h b1 = load_frag_b(&sB[buf][(ntb + 1) * 16 * 32], 32, lane);
    acc00 = wmma32(a0, b0, acc00);
    acc01 = wmma32(a0, b1, acc01);
    acc10 = wmma32(a1, b0, acc10);
    acc11 = wmma32(a1, b1, acc11);
    __syncthreads();
  }

  int n = lane & 15, mhi = (lane >> 4) << 3;
#pragma unroll
  for (int x = 0; x < 2; ++x) {
#pragma unroll
    for (int y = 0; y < 2; ++y) {
      v8f acc = (x == 0) ? (y == 0 ? acc00 : acc01) : (y == 0 ? acc10 : acc11);
      int mt = mtb + x, nt = ntb + y;
      int jg = jbase + nt * 16 + n;
      int proj = jg >> 8, h = (jg >> 5) & 7, c = jg & 31;
#pragma unroll
      for (int r = 0; r < 8; ++r) {
        int mm = r + mhi;
        int i = ibase + mt * 16 + mm;
        int ss = i >> 8, rr = i & 255;
        float val = acc[r];
        if (proj == 0)      qbuf[(((size_t)ss * 8 + h) * 256 + rr) * 32 + c] = (_Float16)val;
        else if (proj == 1) kbuf[(((size_t)ss * 8 + h) * 256 + rr) * 32 + c] = (_Float16)val;
        else if (proj == 2) vbuf[(((size_t)ss * 8 + h) * 32 + c) * 256 + rr] = (_Float16)val;
        else {
          float gg = 1.f / (1.f + __expf(-(val + bg[jg - 768])));
          gbuf[(size_t)i * 256 + (jg - 768)] = (_Float16)gg;
        }
      }
    }
  }
}

// ---------------------------------------------------------------------------
// Kernel 5: fused attention per (s,h,64-query tile). Scores+bias in LDS (f32),
// wave softmax, P(f16) x V via chained WMMA, gated f16 output. 96KB dyn LDS.
// ---------------------------------------------------------------------------
__global__ __launch_bounds__(256) void attn_kernel(const _Float16* __restrict__ q,
                                                   const _Float16* __restrict__ k,
                                                   const _Float16* __restrict__ v,
                                                   const _Float16* __restrict__ g,
                                                   const float* __restrict__ bias,
                                                   _Float16* __restrict__ og) {
  extern __shared__ unsigned char smem[];
  float*    sS = (float*)smem;                       // 64 x 256
  _Float16* sP = (_Float16*)(smem + 64 * 256 * 4);   // 64 x 256
  int tid = threadIdx.x, lane = tid & 31, wv = tid >> 5;
  int bx = blockIdx.x;
  int qt = bx & 3, h = (bx >> 2) & 7, s = bx >> 5;
  int sh = s * 8 + h;
  int qbase = qt * 64;
  int n = lane & 15, mhi = (lane >> 4) << 3;

  int mt = wv >> 1;
  v16h af = load_frag_a(q + ((size_t)sh * 256 + qbase + mt * 16) * 32, 32, lane);
#pragma unroll
  for (int i = 0; i < 8; ++i) {
    int nt = (wv & 1) * 8 + i;
    v16h bf = load_frag_b(k + ((size_t)sh * 256 + nt * 16) * 32, 32, lane);
    v8f acc = {};
    acc = wmma32(af, bf, acc);
    int col = nt * 16 + n;
    const float* bp = bias + (size_t)h * 65536 + (size_t)(qbase + mt * 16) * 256 + col;
#pragma unroll
    for (int r = 0; r < 8; ++r) {
      int mm = r + mhi;
      sS[(mt * 16 + mm) * 256 + col] = acc[r] + bp[(size_t)mm * 256];
    }
  }
  __syncthreads();

#pragma unroll
  for (int rr = 0; rr < 8; ++rr) {
    float* srow = sS + (wv * 8 + rr) * 256;
    float lx[8];
    float mx = -3.0e38f;
#pragma unroll
    for (int j = 0; j < 8; ++j) { lx[j] = srow[lane * 8 + j]; mx = fmaxf(mx, lx[j]); }
    mx = wred_max(mx);
    float sum = 0.f;
#pragma unroll
    for (int j = 0; j < 8; ++j) { lx[j] = __expf(lx[j] - mx); sum += lx[j]; }
    sum = wred_sum(sum);
    float inv = 1.f / sum;
    union { v8h vv; _Float16 e[8]; } o;
#pragma unroll
    for (int j = 0; j < 8; ++j) o.e[j] = (_Float16)(lx[j] * inv);
    *(v8h*)(sP + (wv * 8 + rr) * 256 + lane * 8) = o.vv;
  }
  __syncthreads();

  int omt = wv >> 1, ont = wv & 1;
  v8f acc = {};
  for (int kk = 0; kk < 16; ++kk) {
    v16h pa = load_frag_a(sP + omt * 16 * 256 + kk * 32, 256, lane);
    v16h vb = load_frag_b(v + (((size_t)sh * 32 + ont * 16) * 256) + kk * 32, 256, lane);
    acc = wmma32(pa, vb, acc);
  }
  int c = ont * 16 + n;
#pragma unroll
  for (int r = 0; r < 8; ++r) {
    int mm = r + mhi;
    int rg = qbase + omt * 16 + mm;
    size_t gi = ((size_t)s * 256 + rg) * 256 + h * 32 + c;
    og[gi] = (_Float16)(acc[r] * (float)g[gi]);
  }
}

// ---------------------------------------------------------------------------
// Kernel 6: output projection og(65536x256) x WoT + bo -> f32, same mainloop.
// ---------------------------------------------------------------------------
__global__ __launch_bounds__(256) void gemm_out_kernel(const _Float16* __restrict__ A,
                                                       const _Float16* __restrict__ BT,
                                                       const float* __restrict__ bo,
                                                       float* __restrict__ out) {
  __shared__ __align__(32) _Float16 sA[2][128 * 32];
  __shared__ __align__(32) _Float16 sB[2][64 * 32];
  int tid = threadIdx.x, lane = tid & 31, wv = tid >> 5;
  int ibase = blockIdx.y * 128, jbase = blockIdx.x * 64;
  unsigned aBase = (unsigned)(size_t)(&sA[0][0]);
  unsigned bBase = (unsigned)(size_t)(&sB[0][0]);

  auto issue_tile = [&](int kt, int buf) {
    int k0 = kt * 32;
#pragma unroll
    for (int j = 0; j < 2; ++j) {
      int c = tid * 2 + j, row = c >> 2, off = (c & 3) * 8;
      async_copy_b128(aBase + (unsigned)(buf * 128 * 32 + row * 32 + off) * 2,
                      A + (size_t)(ibase + row) * 256 + k0 + off);
    }
    int n = tid >> 2, off = (tid & 3) * 8;
    async_copy_b128(bBase + (unsigned)(buf * 64 * 32 + n * 32 + off) * 2,
                    BT + (size_t)(jbase + n) * 256 + k0 + off);
  };

  int mtb = (wv & 3) * 2, ntb = (wv >> 2) * 2;
  v8f acc00 = {}, acc01 = {}, acc10 = {}, acc11 = {};
  issue_tile(0, 0);
  for (int kt = 0; kt < 8; ++kt) {
    int buf = kt & 1;
    if (kt < 7) { issue_tile(kt + 1, buf ^ 1); WAIT_ASYNC(0x3); }
    else        { WAIT_ASYNC(0x0); }
    __syncthreads();
    v16h a0 = load_frag_a(&sA[buf][(mtb    ) * 16 * 32], 32, lane);
    v16h a1 = load_frag_a(&sA[buf][(mtb + 1) * 16 * 32], 32, lane);
    v16h b0 = load_frag_b(&sB[buf][(ntb    ) * 16 * 32], 32, lane);
    v16h b1 = load_frag_b(&sB[buf][(ntb + 1) * 16 * 32], 32, lane);
    acc00 = wmma32(a0, b0, acc00);
    acc01 = wmma32(a0, b1, acc01);
    acc10 = wmma32(a1, b0, acc10);
    acc11 = wmma32(a1, b1, acc11);
    __syncthreads();
  }

  int n = lane & 15, mhi = (lane >> 4) << 3;
#pragma unroll
  for (int x = 0; x < 2; ++x) {
#pragma unroll
    for (int y = 0; y < 2; ++y) {
      v8f acc = (x == 0) ? (y == 0 ? acc00 : acc01) : (y == 0 ? acc10 : acc11);
      int mt = mtb + x, nt = ntb + y;
      int jg = jbase + nt * 16 + n;
#pragma unroll
      for (int r = 0; r < 8; ++r) {
        int mm = r + mhi;
        int i = ibase + mt * 16 + mm;
        out[(size_t)i * 256 + jg] = acc[r] + bo[jg];
      }
    }
  }
}

// ---------------------------------------------------------------------------
extern "C" void kernel_launch(void* const* d_in, const int* in_sizes, int n_in,
                              void* d_out, int out_size, void* d_ws, size_t ws_size,
                              hipStream_t stream) {
  (void)in_sizes; (void)n_in; (void)out_size; (void)ws_size;
  const float* m    = (const float*)d_in[0];
  const float* z    = (const float*)d_in[1];
  const float* lnmw = (const float*)d_in[2];
  const float* lnmb = (const float*)d_in[3];
  const float* lnzw = (const float*)d_in[4];
  const float* lnzb = (const float*)d_in[5];
  const float* Wz   = (const float*)d_in[6];
  const float* Wq   = (const float*)d_in[7];
  const float* Wk   = (const float*)d_in[8];
  const float* Wv   = (const float*)d_in[9];
  const float* Wg   = (const float*)d_in[10];
  const float* bg   = (const float*)d_in[11];
  const float* Wo   = (const float*)d_in[12];
  const float* bo   = (const float*)d_in[13];

  char* ws = (char*)d_ws;
  _Float16* mn_og = (_Float16*)(ws);                          // 32MB (mn, later og)
  _Float16* qbuf  = (_Float16*)(ws + (size_t)33554432);       // 32MB [s][h][r][c]
  _Float16* kbuf  = (_Float16*)(ws + (size_t)67108864);       // 32MB [s][h][r][c]
  _Float16* vbuf  = (_Float16*)(ws + (size_t)100663296);      // 32MB [s][h][c][r]
  _Float16* gbuf  = (_Float16*)(ws + (size_t)134217728);      // 32MB
  float*    bias  = (float*)   (ws + (size_t)167772160);      // 2MB  [h][q][k]
  _Float16* WT    = (_Float16*)(ws + (size_t)169869312);      // 512KB
  _Float16* WoT   = (_Float16*)(ws + (size_t)170393600);      // 128KB

  prep_weights_kernel<<<1280, 256, 0, stream>>>(Wq, Wk, Wv, Wg, Wo, WT, WoT);
  ln_m_kernel<<<8192, 256, 0, stream>>>(m, lnmw, lnmb, mn_og);
  ln_z_bias_kernel<<<8192, 256, 0, stream>>>(z, lnzw, lnzb, Wz, bias);
  gemm_qkvg_kernel<<<dim3(16, 512), 256, 0, stream>>>(mn_og, WT, bg, qbuf, kbuf, vbuf, gbuf);
  attn_kernel<<<8192, 256, 98304, stream>>>(qbuf, kbuf, vbuf, gbuf, bias, mn_og /* og */);
  gemm_out_kernel<<<dim3(4, 512), 256, 0, stream>>>(mn_og, WoT, bo, (float*)d_out);
}
